// KANLinear_65652870086721
// MI455X (gfx1250) — matrix operand order repaired and checked by reading, not compiled, not run on hardware
//
#include <hip/hip_runtime.h>

// ---------------------------------------------------------------------------
// KANLinear forward for MI455X (gfx1250, wave32, WMMA + TDM/async LDS DMA).
//
// out[b,o] = (x @ base_weight)[b,o] + sum_{i,k} basis(b,i,k) * W[i,o,k]*scale[o]
//
// spline einsum == GEMM 4096 x (256*66) x 256  (~35 GFLOP, compute bound)
// -> v_wmma_f32_16x16x32_bf16, scaling folded into B.
// Weights pre-scaled/converted to bf16 once into d_ws ([i][o][128], zero pad),
// then streamed to LDS per-iteration with the Tensor Data Mover
// (tensor_load_to_lds, TENSORcnt, hardware LDS padding for bank-conflict-free
// fragment loads), double buffered. Fallbacks: async b128 loads / manual stage.
// ---------------------------------------------------------------------------

typedef __attribute__((ext_vector_type(16))) __bf16 v16bf;
typedef __attribute__((ext_vector_type(8)))  float  v8f;
typedef __attribute__((ext_vector_type(4)))  unsigned int v4u;
typedef __attribute__((ext_vector_type(8)))  int v8i;
typedef __attribute__((ext_vector_type(4)))  int v4i;

#define B_ROWS 4096
#define IN_F   256
#define OUT_F  256
#define NB     66        // number of basis functions
#define KROW   128       // padded K per feature in prep buffer (64 DWORDs/row)
#define STR    104       // bas LDS row stride in halfs
#define STRW   136       // wts LDS row stride in halfs (68 DWORDs: 64 data + 4 TDM pad)
#define ROWS   64        // rows per workgroup
#define OCOLS  128       // output columns per workgroup
#define WBUFH  (OCOLS * STRW)        // halfs per weight LDS buffer (17408)
#define WBUFB  (WBUFH * 2)           // bytes  per weight LDS buffer (34816)

#ifndef __has_builtin
#define __has_builtin(x) 0
#endif
#if __has_builtin(__builtin_amdgcn_tensor_load_to_lds)
#define USE_TDM 1
#else
#define USE_TDM 0
#endif

// round-to-nearest-even f32 -> bf16
__device__ __forceinline__ unsigned short f2bf(float f) {
    unsigned u = __float_as_uint(f);
    u += 0x7FFFu + ((u >> 16) & 1u);
    return (unsigned short)(u >> 16);
}

union FragU { v16bf v; uint4 q[2]; };

// ---- kernel 1: scaling[o] = mean_i max(spline_scaler[i,o], 1e-7) ----------
__global__ void __launch_bounds__(256)
kan_scale_kernel(const float* __restrict__ sc, float* __restrict__ ws) {
    int o = threadIdx.x;
    float s = 0.f;
    for (int i = 0; i < IN_F; ++i)
        s += fmaxf(sc[i * OUT_F + o], 1e-7f);
    ws[o] = s * (1.0f / IN_F);
}

// ---- kernel 1b: pre-scale spline weights -> packed bf16 [i][o][KROW] ------
__global__ void __launch_bounds__(256)
kan_prep_kernel(const float* __restrict__ sw, const float* __restrict__ scale,
                unsigned int* __restrict__ wpre) {
    const int total = IN_F * OUT_F * (KROW / 2);           // dwords (bf16 pairs)
    for (int f = blockIdx.x * 256 + threadIdx.x; f < total; f += gridDim.x * 256) {
        int pair = f % (KROW / 2);
        int io   = f / (KROW / 2);
        int o    = io % OUT_F;
        int k0   = pair * 2;
        float s  = scale[o];
        const float* wrow = sw + (size_t)io * NB;
        unsigned lo = (k0     < NB) ? f2bf(wrow[k0]     * s) : 0u;
        unsigned hi = (k0 + 1 < NB) ? f2bf(wrow[k0 + 1] * s) : 0u;
        wpre[f] = lo | (hi << 16);
    }
}

// ---- TDM issue: one 2D tile [128 rows x 128 bf16] -> LDS with padding -----
#if USE_TDM
__device__ __forceinline__ void tdm_issue(const unsigned short* wpre,
                                          unsigned row0, unsigned lds_addr) {
    unsigned long long gaddr = (unsigned long long)(size_t)wpre
                             + (unsigned long long)row0 * (KROW * 2);
    // D# group0: count=1 | lds_addr | global_addr[56:0] | type=2
    v4u g0 = { 1u,
               lds_addr,
               (unsigned)gaddr,
               (unsigned)((gaddr >> 32) & 0x1FFFFFFu) | (2u << 30) };
    // D# group1: data_size=2B, pad_enable, pad every 64 DW (enc 5) by 4 DW (enc 3),
    //            tensor_dim0=128, tensor_dim1=IN_F*OUT_F, tile 128x128, dim0_stride=128
    v8i g1 = { (int)((1u << 16) | (1u << 20) | (5u << 22) | (3u << 25)),
               (int)(128u << 16),                 // tensor_dim0[15:0] in [63:48]
               0,                                 // dim0[31:16]=0, tensor_dim1[15:0]=0
               (int)(1u | (128u << 16)),          // tensor_dim1[31:16]=1, tile_dim0=128
               128,                               // tile_dim1=128, tile_dim2=0
               128,                               // tensor_dim0_stride[31:0]=128
               0, 0 };                            // stride msbs, tensor_dim1_stride=0
    v4i g2 = { 0, 0, 0, 0 };
    v4i g3 = { 0, 0, 0, 0 };
#if __clang_major__ >= 23
    v8i g4 = { 0, 0, 0, 0, 0, 0, 0, 0 };
    __builtin_amdgcn_tensor_load_to_lds(g0, g1, g2, g3, g4, 0);
#else
    __builtin_amdgcn_tensor_load_to_lds(g0, g1, g2, g3, 0);
#endif
}
#endif

// ---- kernel 2: fused basis + spline GEMM + base GEMM ----------------------
template <bool PRE>
__global__ void __launch_bounds__(256)
kan_main_kernel(const float* __restrict__ xg,
                const float* __restrict__ bwg,
                const float* __restrict__ swg,
                const float* __restrict__ scaleg,
                const unsigned short* __restrict__ wpre,  // bf16 [IN_F][OUT_F][KROW]
                float* __restrict__ outg) {
    extern __shared__ char smem[];
    unsigned short* bas  = (unsigned short*)smem;             // [ROWS x STR]
    unsigned short* wts0 = bas + (size_t)ROWS * STR;          // [2][OCOLS x STRW]
    float* scl  = (float*)(smem + (size_t)ROWS * STR * 2 + 2 * WBUFB);
    float* part = scl + OCOLS;

    const int t   = threadIdx.x;
    const int rb  = (blockIdx.x >> 1) * ROWS;
    const int o0  = (blockIdx.x & 1) * OCOLS;

    const int lane  = t & 31;
    const int w     = t >> 5;
    const int n16   = lane & 15;
    const int kh    = lane >> 4;
    const int rslab = (w & 3) * 16;
    const int otb   = (w >> 2) * 4;

    // zero all bf16 tile buffers once (K padding regions stay zero)
    for (int f = t; f < (ROWS * STR * 2 + 2 * WBUFB) / 4; f += 256)
        ((unsigned int*)smem)[f] = 0u;
    if (!PRE && t < OCOLS) scl[t] = scaleg[o0 + t];
    __syncthreads();

    const unsigned wts_lds0 = (unsigned)(size_t)wts0;  // LDS byte address (low 32 bits)

#if !USE_TDM
    int dma_o[6], dma_kk[6];
#endif
    if (PRE) {
#if USE_TDM
        if (w == 0) tdm_issue(wpre, (unsigned)(0 * OUT_F + o0), wts_lds0);
#else
        // async-load fallback: 12 useful b128 chunks per row (k<96), 6/thread
        #pragma unroll
        for (int j = 0; j < 6; ++j) {
            int c = t + j * 256;
            dma_o[j]  = c / 12;
            dma_kk[j] = (c % 12) * 8;
        }
        #pragma unroll
        for (int j = 0; j < 6; ++j) {
            unsigned lds_off = wts_lds0 + (unsigned)(dma_o[j] * (STRW * 2) + dma_kk[j] * 2);
            unsigned voff    = (unsigned)(((0 * OUT_F + o0 + dma_o[j]) * KROW + dma_kk[j]) * 2);
            asm volatile("global_load_async_to_lds_b128 %0, %1, %2"
                         :: "v"(lds_off), "v"(voff), "s"(wpre) : "memory");
        }
#endif
    }

    v8f acc[4] = {};

    // ================= spline GEMM: loop over input features ===============
    for (int i = 0; i < IN_F; ++i) {
        const int pbuf = PRE ? (i & 1) : 0;

        if (PRE && i + 1 < IN_F) {
            unsigned bbase = wts_lds0 + (unsigned)(((i + 1) & 1) * WBUFB);
#if USE_TDM
            if (w == 0) tdm_issue(wpre, (unsigned)((i + 1) * OUT_F + o0), bbase);
#else
            #pragma unroll
            for (int j = 0; j < 6; ++j) {
                unsigned lds_off = bbase + (unsigned)(dma_o[j] * (STRW * 2) + dma_kk[j] * 2);
                unsigned voff    = (unsigned)((((i + 1) * OUT_F + o0 + dma_o[j]) * KROW + dma_kk[j]) * 2);
                asm volatile("global_load_async_to_lds_b128 %0, %1, %2"
                             :: "v"(lds_off), "v"(voff), "s"(wpre) : "memory");
            }
#endif
        } else if (!PRE) {
            // manual staging: spline_weight[i, o0:o0+128, :] * scale[o] -> bf16
            for (int f = t; f < OCOLS * NB; f += 256) {
                int o = f / NB;
                int k = f - o * NB;
                float wv = swg[(size_t)i * OUT_F * NB + (size_t)(o0 + o) * NB + k] * scl[o];
                wts0[o * STRW + k] = f2bf(wv);
            }
        }

        // --- basis row for x[rb+r, i] (4 threads per row, 17 centers each)
        const int r = t >> 2, h = t & 3;
        float xv = xg[(size_t)(rb + r) * IN_F + i];
        float xc = fminf(fmaxf(xv, -1.f), 1.f);
        float xs = (xc + 1.f) * (63.f / (2.f + 1e-7f));
        float gi = fminf(fmaxf(floorf(xs), 0.f), 61.f);
        float lx = xs - gi;
        float vb[17];
        float s = 0.f;
        #pragma unroll
        for (int j = 0; j < 17; ++j) {
            int k = h * 17 + j;
            float d = lx - (float)k * (1.f / 65.f);
            float e = (k < NB) ? __expf(-2178.f * d * d) : 0.f;
            vb[j] = e; s += e;
        }
        part[t] = s;
        __syncthreads();

        int tb = t & ~3;
        float inv = 1.f / (part[tb] + part[tb + 1] + part[tb + 2] + part[tb + 3] + 1e-7f);
        #pragma unroll
        for (int j = 0; j < 17; ++j) {
            int k = h * 17 + j;
            if (k < NB) bas[r * STR + k] = f2bf(vb[j] * inv);
        }

        if (PRE) {
            // current buffer's DMA is the OLDER group (in-order completion):
            // wait until only the newly issued group remains.
#if USE_TDM
            if (i + 1 < IN_F) __builtin_amdgcn_s_wait_tensorcnt(1);
            else              __builtin_amdgcn_s_wait_tensorcnt(0);
#else
            if (i + 1 < IN_F) asm volatile("s_wait_asynccnt 0x6" ::: "memory");
            else              asm volatile("s_wait_asynccnt 0x0" ::: "memory");
#endif
        }
        __syncthreads();

        // --- WMMA: [64 x 96] x [96 x 128]
        const unsigned short* wcur = wts0 + pbuf * WBUFH;
        #pragma unroll
        for (int kc = 0; kc < 3; ++kc) {
            FragU A;
            const unsigned short* ap = bas + (rslab + n16) * STR + kc * 32 + kh * 16;
            A.q[0] = *(const uint4*)ap;
            A.q[1] = *(const uint4*)(ap + 8);
            FragU Bf[4];
            #pragma unroll
            for (int o = 0; o < 4; ++o) {
                const unsigned short* bp = wcur + ((otb + o) * 16 + n16) * STRW + kc * 32 + kh * 16;
                Bf[o].q[0] = *(const uint4*)bp;
                Bf[o].q[1] = *(const uint4*)(bp + 8);
            }
            #pragma unroll
            for (int o = 0; o < 4; ++o)
                acc[o] = __builtin_amdgcn_wmma_f32_16x16x32_bf16(
                    false, A.v, false, Bf[o].v, (short)0, acc[o], false, false);
        }
        __syncthreads();
    }

    // ================= base GEMM: x @ base_weight, K=256 in 8 chunks =======
    #pragma unroll 1
    for (int kc2 = 0; kc2 < 8; ++kc2) {
        for (int f = t; f < ROWS * 32; f += 256) {
            int r = f >> 5, kk = f & 31;
            bas[r * STR + kk] = f2bf(xg[(size_t)(rb + r) * IN_F + kc2 * 32 + kk]);
        }
        for (int f = t; f < 32 * OCOLS; f += 256) {
            int k = f >> 7, o = f & 127;
            wts0[o * STRW + k] = f2bf(bwg[(size_t)(kc2 * 32 + k) * OUT_F + o0 + o]);
        }
        __syncthreads();

        FragU A;
        const unsigned short* ap = bas + (rslab + n16) * STR + kh * 16;
        A.q[0] = *(const uint4*)ap;
        A.q[1] = *(const uint4*)(ap + 8);
        FragU Bf[4];
        #pragma unroll
        for (int o = 0; o < 4; ++o) {
            const unsigned short* bp = wts0 + ((otb + o) * 16 + n16) * STRW + kh * 16;
            Bf[o].q[0] = *(const uint4*)bp;
            Bf[o].q[1] = *(const uint4*)(bp + 8);
        }
        #pragma unroll
        for (int o = 0; o < 4; ++o)
            acc[o] = __builtin_amdgcn_wmma_f32_16x16x32_bf16(
                false, A.v, false, Bf[o].v, (short)0, acc[o], false, false);
        __syncthreads();
    }

    // ================= store ===============================================
    #pragma unroll
    for (int o = 0; o < 4; ++o) {
        int col = o0 + (otb + o) * 16 + n16;
        #pragma unroll
        for (int vg = 0; vg < 8; ++vg) {
            int row = rb + rslab + kh * 8 + vg;
            outg[(size_t)row * OUT_F + col] = acc[o][vg];
        }
    }
}

// ---------------------------------------------------------------------------
extern "C" void kernel_launch(void* const* d_in, const int* in_sizes, int n_in,
                              void* d_out, int out_size, void* d_ws, size_t ws_size,
                              hipStream_t stream) {
    const float* x  = (const float*)d_in[0];   // [4096, 256]
    const float* bw = (const float*)d_in[1];   // [256, 256]
    const float* sw = (const float*)d_in[2];   // [256, 256, 66]
    const float* sc = (const float*)d_in[3];   // [256, 256]
    float* out = (float*)d_out;                // [4096, 256]
    float* ws  = (float*)d_ws;

    dim3 grid((B_ROWS / ROWS) * (OUT_F / OCOLS));            // 128 blocks
    size_t smem = (size_t)ROWS * STR * 2 + 2 * WBUFB         // bf16 tile buffers
                + (size_t)OCOLS * 4 + 256 * 4;               // scale + partial sums

    const size_t pre_bytes = (size_t)IN_F * OUT_F * KROW * 2;  // 16.8 MB bf16 weights
    kan_scale_kernel<<<1, 256, 0, stream>>>(sc, ws);           // scaling -> ws[0..255]

    if (ws_size >= 1024 + pre_bytes) {
        unsigned short* wpre = (unsigned short*)((char*)d_ws + 1024);
        kan_prep_kernel<<<4096, 256, 0, stream>>>(sw, ws, (unsigned int*)wpre);
        kan_main_kernel<true><<<grid, 256, smem, stream>>>(x, bw, sw, ws, wpre, out);
    } else {
        kan_main_kernel<false><<<grid, 256, smem, stream>>>(x, bw, sw, ws, nullptr, out);
    }
}
